// ScaledDotProduct_55087250538570
// MI455X (gfx1250) — compile-verified, matrix operand order checked by reading, not run on hardware
//
#include <hip/hip_runtime.h>
#include <math.h>

typedef __attribute__((ext_vector_type(16))) _Float16 v16h;
typedef __attribute__((ext_vector_type(8)))  float    v8f;
typedef __attribute__((ext_vector_type(4)))  float    f4;

namespace {
constexpr int Bb = 4, Hh = 16, Ss = 2048, Dd = 64;
constexpr float kScale = 0.125f;   // 1/sqrt(64)
constexpr int QTILE = 128;         // q rows per block = 8 waves * 16
constexpr int KT    = 64;          // kv rows per tile
constexpr int NT    = Ss / KT;     // 32 kv tiles
}

// 16-bit WMMA A/B fragment K-index (ISA 7.12.2):
// element i of lane L -> K = 16*(i>>3) + 8*(L>=16) + 2*((i>>1)&3) + (i&1)
//                          = 16*(i>>3) + 8*(L>=16) + (i&7)
// Consecutive i give contiguous 8-half runs -> LDS reads vectorize to b128.
__device__ __forceinline__ int kidx(int i, int lh) {
  return ((i >> 3) << 4) + (lh << 3) + (((i >> 1) & 3) << 1) + (i & 1);
}

// S^T_g += K[16g..16g+16) @ Q^T  (A = K-frag from LDS, B = Q^T-frag in regs)
#define ST_STEP(sacc, g) do {                                                             \
    v16h a_;                                                                              \
    _Pragma("unroll") for (int i_ = 0; i_ < 16; ++i_)                                     \
      a_[i_] = Ksh[(16*(g) + ln) * Dd + kidx(i_, lh)];                                    \
    sacc = __builtin_amdgcn_wmma_f32_16x16x32_f16(false, a_, false, bq0, (short)0, sacc, false, false); \
    _Pragma("unroll") for (int i_ = 0; i_ < 16; ++i_)                                     \
      a_[i_] = Ksh[(16*(g) + ln) * Dd + 32 + kidx(i_, lh)];                               \
    sacc = __builtin_amdgcn_wmma_f32_16x16x32_f16(false, a_, false, bq1, (short)0, sacc, false, false); \
  } while (0)

// O^T_c += V^T[16c..16c+16, :] @ P^T  (A = V^T-frag from LDS, B = P^T in regs)
#define OT_STEP(oacc, c) do {                                                             \
    v16h a_;                                                                              \
    _Pragma("unroll") for (int i_ = 0; i_ < 16; ++i_)                                     \
      a_[i_] = Vts[(16*(c) + ln) * KT + kidx(i_, lh)];                                    \
    oacc = __builtin_amdgcn_wmma_f32_16x16x32_f16(false, a_, false, bp0, (short)0, oacc, false, false); \
    _Pragma("unroll") for (int i_ = 0; i_ < 16; ++i_)                                     \
      a_[i_] = Vts[(16*(c) + ln) * KT + 32 + kidx(i_, lh)];                               \
    oacc = __builtin_amdgcn_wmma_f32_16x16x32_f16(false, a_, false, bp1, (short)0, oacc, false, false); \
  } while (0)

__global__ __launch_bounds__(256, 2)
void fa_fwd_wmma(const float* __restrict__ q, const float* __restrict__ k,
                 const float* __restrict__ v, float* __restrict__ o) {
  __shared__ _Float16 Ksh[KT * Dd];        // K tile  [k][d]   8 KB
  __shared__ _Float16 Vts[Dd * KT];        // V tile  [d][k]   8 KB (transposed)

  const int tid  = threadIdx.x;
  const int lane = tid & 31;
  const int w    = tid >> 5;
  const int lh   = lane >> 4;
  const int ln   = lane & 15;

  const int qblocks = Ss / QTILE;
  const int bh    = blockIdx.x / qblocks;
  const int qblk  = blockIdx.x % qblocks;
  const int qbase = qblk * QTILE + w * 16;

  const float* qp = q + ((size_t)bh * Ss + qbase) * Dd;
  const float* kp = k + (size_t)bh * Ss * Dd;
  const float* vp = v + (size_t)bh * Ss * Dd;
  float*       op = o + ((size_t)bh * Ss + qbase) * Dd;

  // ---- Q tile as two B-fragments (Q^T), pre-scaled by 1/sqrt(D) ----
  v16h bq0, bq1;
#pragma unroll
  for (int i = 0; i < 16; ++i) {
    const int kk = kidx(i, lh);
    bq0[i] = (_Float16)(qp[ln * Dd + kk] * kScale);
    bq1[i] = (_Float16)(qp[ln * Dd + 32 + kk] * kScale);
  }

  // constant all-ones A fragment: ones @ P^T = column sums (softmax denominator)
  v16h aones;
#pragma unroll
  for (int i = 0; i < 16; ++i) aones[i] = (_Float16)1.0f;

  // O^T accumulators: lane = q-col (ln), VGPR i = d row 16c + i + 8*lh
  v8f o0 = {}, o1 = {}, o2 = {}, o3 = {};
  v8f lacc = {};               // denominator (every VGPR row holds q-col sum)
  float mrow = -INFINITY;      // running max for this lane's q-column

  const int kvrow = tid >> 2;            // V staging: this thread's k row
  const int d0    = (tid & 3) << 4;      // and d column base (16 cols)

  // ---- register double buffer: preload tile 0 ----
  f4 krg[4], vrg[4];
  {
    const f4* kt0 = (const f4*)(kp + (size_t)tid * 16);
    const f4* vt0 = (const f4*)(vp + (size_t)tid * 16);
#pragma unroll
    for (int j = 0; j < 4; ++j) { krg[j] = kt0[j]; vrg[j] = vt0[j]; }
  }

  for (int t = 0; t < NT; ++t) {
    __syncthreads();                     // previous tile's readers done
    // commit staged registers to LDS (fp32 -> f16); V goes in transposed
    {
      _Float16* kd = Ksh + tid * 16;     // contiguous -> merged ds_store
#pragma unroll
      for (int j = 0; j < 4; ++j)
#pragma unroll
        for (int c = 0; c < 4; ++c) kd[4 * j + c] = (_Float16)krg[j][c];
#pragma unroll
      for (int j = 0; j < 4; ++j)
#pragma unroll
        for (int c = 0; c < 4; ++c)
          Vts[(d0 + 4 * j + c) * KT + kvrow] = (_Float16)vrg[j][c];
    }
    if (t + 1 < NT) {                    // preload next tile -> overlaps compute
      const f4* ktn = (const f4*)(kp + (size_t)(t + 1) * KT * Dd + tid * 16);
      const f4* vtn = (const f4*)(vp + (size_t)(t + 1) * KT * Dd + tid * 16);
#pragma unroll
      for (int j = 0; j < 4; ++j) { krg[j] = ktn[j]; vrg[j] = vtn[j]; }
    }
    __syncthreads();

    // ---- S^T = K_tile @ (Q*scale)^T : four 16(key)x16(q) accumulators.
    // Lane holds q-col = ln; VGPR j = key 16g + j + 8*lh. ----
    v8f s0 = {}, s1 = {}, s2 = {}, s3 = {};
    ST_STEP(s0, 0); ST_STEP(s1, 1); ST_STEP(s2, 2); ST_STEP(s3, 3);

    // ---- row max for this lane's q-column: in-lane tree + ONE xor-16 bpermute ----
    float mloc = s0[0];
#pragma unroll
    for (int i = 0; i < 8; ++i) {
      mloc = fmaxf(mloc, s0[i]); mloc = fmaxf(mloc, s1[i]);
      mloc = fmaxf(mloc, s2[i]); mloc = fmaxf(mloc, s3[i]);
    }
    mloc = fmaxf(mloc, __shfl_xor(mloc, 16, 32));   // other half holds keys+8
    const float mnew = fmaxf(mrow, mloc);
    const float fac  = __expf(mrow - mnew);
    mrow = mnew;

    // ---- P^T fragments built IN REGISTERS (S^T C-layout == B-operand layout):
    // bp0 element i -> key 16*(i>>3)+8*lh+(i&7) = tile (i>>3), VGPR (i&7), same lane ----
    v16h bp0, bp1;
#pragma unroll
    for (int i = 0; i < 8; ++i) {
      bp0[i]     = (_Float16)__expf(s0[i] - mnew);
      bp0[i + 8] = (_Float16)__expf(s1[i] - mnew);
      bp1[i]     = (_Float16)__expf(s2[i] - mnew);
      bp1[i + 8] = (_Float16)__expf(s3[i] - mnew);
    }

    // ---- rescale running state by fac ----
#pragma unroll
    for (int i = 0; i < 8; ++i) {
      o0[i] *= fac; o1[i] *= fac; o2[i] *= fac; o3[i] *= fac; lacc[i] *= fac;
    }

    // ---- denominator: lacc += ones @ P^T (column sums, all keys covered) ----
    lacc = __builtin_amdgcn_wmma_f32_16x16x32_f16(false, aones, false, bp0, (short)0, lacc, false, false);
    lacc = __builtin_amdgcn_wmma_f32_16x16x32_f16(false, aones, false, bp1, (short)0, lacc, false, false);

    // ---- O^T += V^T @ P^T : four d-row chunks, K-dim 64 = 2 WMMAs each ----
    OT_STEP(o0, 0); OT_STEP(o1, 1); OT_STEP(o2, 2); OT_STEP(o3, 3);
  }

  // ---- normalize and write out: lane ln owns q-row ln; per-lane runs of 8
  // consecutive d values -> merges into global_store_b128 ----
  const float invl = 1.0f / lacc[0];
  float* orow = op + ln * Dd + (lh << 3);
#pragma unroll
  for (int i = 0; i < 8; ++i) {
    orow[i]      = o0[i] * invl;
    orow[16 + i] = o1[i] * invl;
    orow[32 + i] = o2[i] * invl;
    orow[48 + i] = o3[i] * invl;
  }
}

extern "C" void kernel_launch(void* const* d_in, const int* in_sizes, int n_in,
                              void* d_out, int out_size, void* d_ws, size_t ws_size,
                              hipStream_t stream) {
  (void)in_sizes; (void)n_in; (void)out_size; (void)d_ws; (void)ws_size;
  const float* q = (const float*)d_in[0];
  const float* k = (const float*)d_in[1];
  const float* v = (const float*)d_in[2];
  float* o = (float*)d_out;
  const int blocks = Bb * Hh * (Ss / QTILE);   // 1024
  fa_fwd_wmma<<<blocks, 256, 0, stream>>>(q, k, v, o);
}
